// Text2ImgCA_26980984553969
// MI455X (gfx1250) — matrix operand order repaired and checked by reading, not compiled, not run on hardware
//
#include <hip/hip_runtime.h>
#include <hip/hip_bf16.h>

// ---------------------------------------------------------------------------
// Text2Img cross-attention for MI455X (gfx1250), wave32 + WMMA f16->f32.
//
// All six GEMMs run through v_wmma_f32_16x16x32_f16. GEMMs are phrased as
// C = A * Bt^T with Bt row-major [N,K] so every WMMA fragment load is a
// contiguous global b128. Block = 256 threads = 8 waves; each wave owns a
// 16x64 output strip (A-fragment reused across 4 WMMAs per 32-wide K step).
// The K loop is software-pipelined (depth-1 register double buffering) so
// next-step global_load_b128s overlap with the current step's 4 WMMAs
// instead of serializing on s_wait_loadcnt 0.
// ---------------------------------------------------------------------------

typedef __attribute__((ext_vector_type(16))) _Float16 v16h;
typedef __attribute__((ext_vector_type(8)))  _Float16 half8;
typedef __attribute__((ext_vector_type(8)))  float    v8f;

#define BATCH   32
#define NTXT    256
#define MPIX    1024
#define TEXTC   768
#define IMGC    1024
#define DQKC    512

// ---------------------------------------------------------------------------
// f32 -> f16 elementwise convert (grid-stride)
// ---------------------------------------------------------------------------
__global__ __launch_bounds__(256)
void cvt_f32_f16_kernel(const float* __restrict__ in, _Float16* __restrict__ out, long n) {
    long i = (long)blockIdx.x * blockDim.x + threadIdx.x;
    long stride = (long)gridDim.x * blockDim.x;
    for (; i < n; i += stride) out[i] = (_Float16)in[i];
}

// ---------------------------------------------------------------------------
// W[K,N] f32 -> Wt[N,K] f16  (weights are tiny; simple elementwise)
// ---------------------------------------------------------------------------
__global__ __launch_bounds__(256)
void transpose_cvt_kernel(const float* __restrict__ W, _Float16* __restrict__ Wt,
                          int K, int N) {
    long total = (long)K * N;
    long i = (long)blockIdx.x * blockDim.x + threadIdx.x;
    long stride = (long)gridDim.x * blockDim.x;
    for (; i < total; i += stride) {
        int k = (int)(i / N);
        int n = (int)(i % N);
        Wt[(long)n * K + k] = (_Float16)W[i];
    }
}

// ---------------------------------------------------------------------------
// Tiled WMMA GEMM:  C[M,N] = A[M,K] * Bt[N,K]^T (+bias) (+ELU)
//   OMODE 0: f16 output, row-major ldc
//   OMODE 1: f16 output, transposed-batched store (for v^T):
//              dst[(row>>tShift)*tPlane + col*tLd + (row & tMask)]
//   OMODE 2: f32 output, row-major ldc
// Block tile 64(M) x 128(N); 8 waves of 32; wave tile 16x64; K step 32.
// ---------------------------------------------------------------------------
template<int OMODE, bool HAS_BIAS, bool DO_ELU>
__global__ __launch_bounds__(256)
void gemm_bt_kernel(const _Float16* __restrict__ A,
                    const _Float16* __restrict__ Bt,
                    const float*    __restrict__ bias,
                    void*           __restrict__ Cout,
                    int K, int lda, int ldb, int ldc,
                    long aBatch, long bBatch, long cBatch,
                    int tShift, int tMask, int tLd, long tPlane)
{
    const int lane  = threadIdx.x & 31;
    const int wave  = threadIdx.x >> 5;
    const int waveM = wave & 3;   // 4 wave rows
    const int waveN = wave >> 2;  // 2 wave col-groups
    const long batch = blockIdx.z;

    const _Float16* Ab  = A  + batch * aBatch;
    const _Float16* Btb = Bt + batch * bBatch;

    const int rowTile = blockIdx.y * 64 + waveM * 16;
    const int colBase = blockIdx.x * 128 + waveN * 64;

    // A fragment (16x32 f16): lane L holds row (L&15);
    // lanes 0-15: K {0..7,16..23}; lanes 16-31: K {8..15,24..31}
    const int aRow = rowTile + (lane & 15);
    const int kLoA = (lane < 16) ? 0 : 8;
    // B fragment (32x16): lane holds column (lane&15);
    // lanes 0-15: K {0..7,16..23}; lanes 16-31: K {8..15,24..31}
    const int kOffB = (lane < 16) ? 0 : 8;
    const int bCol  = lane & 15;

    // Hoisted per-lane base pointers (K loop only adds an immediate).
    const _Float16* aPtr = Ab + (long)aRow * lda + kLoA;
    const _Float16* bP0  = Btb + (long)(colBase + bCol) * ldb + kOffB;
    const long bTS = (long)16 * ldb;   // per output-tile stride within the strip

    auto loadA = [&](int k0) -> v16h {
        half8 lo = *(const half8*)(aPtr + k0);
        half8 hi = *(const half8*)(aPtr + k0 + 16);
        return __builtin_shufflevector(lo, hi,
                   0,1,2,3,4,5,6,7,8,9,10,11,12,13,14,15);
    };
    auto loadB = [&](int t, int k0) -> v16h {
        const _Float16* p = bP0 + (long)t * bTS + k0;
        half8 lo = *(const half8*)(p);
        half8 hi = *(const half8*)(p + 16);
        return __builtin_shufflevector(lo, hi,
                   0,1,2,3,4,5,6,7,8,9,10,11,12,13,14,15);
    };

    v8f acc[4] = {};

    // ---- software-pipelined K loop (depth-1 double buffering) ----
    v16h aCur = loadA(0);
    v16h bCur0 = loadB(0, 0), bCur1 = loadB(1, 0),
         bCur2 = loadB(2, 0), bCur3 = loadB(3, 0);

    for (int k0 = 0; k0 + 32 < K; k0 += 32) {
        // prefetch next K step while WMMAs below execute
        v16h aNxt  = loadA(k0 + 32);
        v16h bNxt0 = loadB(0, k0 + 32);
        v16h bNxt1 = loadB(1, k0 + 32);
        v16h bNxt2 = loadB(2, k0 + 32);
        v16h bNxt3 = loadB(3, k0 + 32);

        acc[0] = __builtin_amdgcn_wmma_f32_16x16x32_f16(false, aCur, false, bCur0,
                                                        (short)0, acc[0], false, false);
        acc[1] = __builtin_amdgcn_wmma_f32_16x16x32_f16(false, aCur, false, bCur1,
                                                        (short)0, acc[1], false, false);
        acc[2] = __builtin_amdgcn_wmma_f32_16x16x32_f16(false, aCur, false, bCur2,
                                                        (short)0, acc[2], false, false);
        acc[3] = __builtin_amdgcn_wmma_f32_16x16x32_f16(false, aCur, false, bCur3,
                                                        (short)0, acc[3], false, false);

        aCur = aNxt;
        bCur0 = bNxt0; bCur1 = bNxt1; bCur2 = bNxt2; bCur3 = bNxt3;
    }
    // tail step
    acc[0] = __builtin_amdgcn_wmma_f32_16x16x32_f16(false, aCur, false, bCur0,
                                                    (short)0, acc[0], false, false);
    acc[1] = __builtin_amdgcn_wmma_f32_16x16x32_f16(false, aCur, false, bCur1,
                                                    (short)0, acc[1], false, false);
    acc[2] = __builtin_amdgcn_wmma_f32_16x16x32_f16(false, aCur, false, bCur2,
                                                    (short)0, acc[2], false, false);
    acc[3] = __builtin_amdgcn_wmma_f32_16x16x32_f16(false, aCur, false, bCur3,
                                                    (short)0, acc[3], false, false);

    // C layout (f32 16x16): col = lane&15; VGPR r -> row (lane<16 ? r : 8+r)
    const int cCol  = lane & 15;
    const int cRow0 = rowTile + ((lane < 16) ? 0 : 8);
#pragma unroll
    for (int t = 0; t < 4; ++t) {
        const int col = colBase + t * 16 + cCol;
        const float bv = HAS_BIAS ? bias[col] : 0.0f;
#pragma unroll
        for (int r = 0; r < 8; ++r) {
            float x = acc[t][r] + bv;
            if (DO_ELU) x = (x > 0.0f) ? x : (__expf(x) - 1.0f);
            const int row = cRow0 + r;
            if (OMODE == 2) {
                ((float*)Cout)[cBatch * batch + (long)row * ldc + col] = x;
            } else if (OMODE == 0) {
                ((_Float16*)Cout)[cBatch * batch + (long)row * ldc + col] =
                    (_Float16)x;
            } else { // OMODE == 1: transposed-batched (v^T)
                const long bi = (long)(row >> tShift);
                ((_Float16*)Cout)[bi * tPlane + (long)col * tLd + (row & tMask)] =
                    (_Float16)x;
            }
        }
    }
}

// ---------------------------------------------------------------------------
// In-place row softmax over 1024 elems + post-softmax mask (NO renorm).
// One 256-thread block per (b,n) row. Writes f32 beta (in place, into d_out)
// and an f16 copy for the beta @ v WMMA.
// ---------------------------------------------------------------------------
__global__ __launch_bounds__(256)
void softmax_mask_kernel(float* __restrict__ sBeta,
                         const float* __restrict__ masks,
                         _Float16* __restrict__ betah)
{
    const long row = blockIdx.x;
    float*       sp = sBeta + row * MPIX;
    const float* mp = masks + row * MPIX;
    _Float16*    hp = betah + row * MPIX;

    __shared__ float red[256];
    const int t = threadIdx.x;

    float x[4];
#pragma unroll
    for (int i = 0; i < 4; ++i) x[i] = sp[t + 256 * i];

    float mx = fmaxf(fmaxf(x[0], x[1]), fmaxf(x[2], x[3]));
    red[t] = mx; __syncthreads();
    for (int s = 128; s > 0; s >>= 1) {
        if (t < s) red[t] = fmaxf(red[t], red[t + s]);
        __syncthreads();
    }
    mx = red[0]; __syncthreads();

    float sum = 0.0f;
#pragma unroll
    for (int i = 0; i < 4; ++i) { x[i] = __expf(x[i] - mx); sum += x[i]; }
    red[t] = sum; __syncthreads();
    for (int s = 128; s > 0; s >>= 1) {
        if (t < s) red[t] += red[t + s];
        __syncthreads();
    }
    const float inv = 1.0f / red[0];

#pragma unroll
    for (int i = 0; i < 4; ++i) {
        const float b = x[i] * inv * mp[t + 256 * i];
        sp[t + 256 * i] = b;
        hp[t + 256 * i] = (_Float16)b;
    }
}

// ---------------------------------------------------------------------------
// Host-side orchestration
// ---------------------------------------------------------------------------
extern "C" void kernel_launch(void* const* d_in, const int* in_sizes, int n_in,
                              void* d_out, int out_size, void* d_ws, size_t ws_size,
                              hipStream_t stream) {
    (void)in_sizes; (void)n_in; (void)out_size; (void)ws_size;

    const float* x1    = (const float*)d_in[0];   // [32,256,768]
    const float* x2    = (const float*)d_in[1];   // [32,32,32,1024]
    const float* masks = (const float*)d_in[2];   // [32,256,1024]
    const float* Wq    = (const float*)d_in[3];   // [768,512]
    const float* bq    = (const float*)d_in[4];
    const float* Wk    = (const float*)d_in[5];   // [1024,512]
    const float* bk    = (const float*)d_in[6];
    const float* Wv    = (const float*)d_in[7];   // [1024,768]
    const float* bv    = (const float*)d_in[8];
    const float* Wo    = (const float*)d_in[9];   // [768,768]
    const float* bo    = (const float*)d_in[10];

    float* oOut    = (float*)d_out;                               // [32,256,768]
    float* betaOut = oOut + (size_t)BATCH * NTXT * TEXTC;         // [32,256,1024]

    // ---- workspace carve-up (f16 intermediates), 256B aligned ----
    char* ws = (char*)d_ws;
    size_t off = 0;
    auto carve = [&](size_t bytes) -> char* {
        char* p = ws + off;
        off = (off + bytes + 255) & ~(size_t)255;
        return p;
    };
    const long nX1 = (long)BATCH * NTXT * TEXTC;     // 6.29M
    const long nX2 = (long)BATCH * MPIX * IMGC;      // 33.55M
    _Float16* x1h   = (_Float16*)carve(nX1 * 2);
    _Float16* x2h   = (_Float16*)carve(nX2 * 2);
    _Float16* WqT   = (_Float16*)carve((size_t)DQKC * TEXTC * 2);   // [512,768]
    _Float16* WkT   = (_Float16*)carve((size_t)DQKC * IMGC * 2);    // [512,1024]
    _Float16* WvT   = (_Float16*)carve((size_t)TEXTC * IMGC * 2);   // [768,1024]
    _Float16* WoT   = (_Float16*)carve((size_t)TEXTC * TEXTC * 2);  // [768,768]
    _Float16* qh    = (_Float16*)carve((size_t)BATCH * NTXT * DQKC * 2);
    _Float16* kh    = (_Float16*)carve((size_t)BATCH * MPIX * DQKC * 2);
    _Float16* vT    = (_Float16*)carve((size_t)BATCH * TEXTC * MPIX * 2); // [b][768][1024]
    _Float16* betah = (_Float16*)carve((size_t)BATCH * NTXT * MPIX * 2);
    _Float16* o1h   = (_Float16*)carve((size_t)BATCH * NTXT * TEXTC * 2);

    // ---- 1. casts / weight transposes ----
    cvt_f32_f16_kernel<<<4096, 256, 0, stream>>>(x1, x1h, nX1);
    cvt_f32_f16_kernel<<<8192, 256, 0, stream>>>(x2, x2h, nX2);
    transpose_cvt_kernel<<<1536, 256, 0, stream>>>(Wq, WqT, TEXTC, DQKC);
    transpose_cvt_kernel<<<2048, 256, 0, stream>>>(Wk, WkT, IMGC, DQKC);
    transpose_cvt_kernel<<<3072, 256, 0, stream>>>(Wv, WvT, IMGC, TEXTC);
    transpose_cvt_kernel<<<2304, 256, 0, stream>>>(Wo, WoT, TEXTC, TEXTC);

    const dim3 blk(256);

    // ---- 2. q = x1 @ Wq + bq  -> f16 [8192,512] ----
    gemm_bt_kernel<0, true, false><<<dim3(DQKC / 128, (BATCH * NTXT) / 64, 1), blk, 0, stream>>>(
        x1h, WqT, bq, qh, TEXTC, TEXTC, TEXTC, DQKC, 0, 0, 0, 0, 0, 0, 0);

    // ---- 3. k = x2 @ Wk + bk  -> f16 [32768,512] ----
    gemm_bt_kernel<0, true, false><<<dim3(DQKC / 128, (BATCH * MPIX) / 64, 1), blk, 0, stream>>>(
        x2h, WkT, bk, kh, IMGC, IMGC, IMGC, DQKC, 0, 0, 0, 0, 0, 0, 0);

    // ---- 4. v = x2 @ Wv + bv -> f16 transposed per batch: vT[b][768][1024] ----
    gemm_bt_kernel<1, true, false><<<dim3(TEXTC / 128, (BATCH * MPIX) / 64, 1), blk, 0, stream>>>(
        x2h, WvT, bv, vT, IMGC, IMGC, IMGC, 0, 0, 0, 0,
        /*tShift=*/10, /*tMask=*/MPIX - 1, /*tLd=*/MPIX, /*tPlane=*/(long)TEXTC * MPIX);

    // ---- 5. s = q @ k^T -> f32 straight into d_out beta region (batched) ----
    gemm_bt_kernel<2, false, false><<<dim3(MPIX / 128, NTXT / 64, BATCH), blk, 0, stream>>>(
        qh, kh, nullptr, betaOut, DQKC, DQKC, DQKC, MPIX,
        (long)NTXT * DQKC, (long)MPIX * DQKC, (long)NTXT * MPIX, 0, 0, 0, 0);

    // ---- 6. softmax rows (in place) + post-softmax mask; f16 copy ----
    softmax_mask_kernel<<<BATCH * NTXT, 256, 0, stream>>>(betaOut, masks, betah);

    // ---- 7. o1 = beta @ v -> f16 [b][256][768] (batched, Bt = vT[b]) ----
    gemm_bt_kernel<0, false, false><<<dim3(TEXTC / 128, NTXT / 64, BATCH), blk, 0, stream>>>(
        betah, vT, nullptr, o1h, MPIX, MPIX, MPIX, TEXTC,
        (long)NTXT * MPIX, (long)TEXTC * MPIX, (long)NTXT * TEXTC, 0, 0, 0, 0);

    // ---- 8. o = elu(o1 @ Wo + bo) -> f32 d_out ----
    gemm_bt_kernel<2, true, true><<<dim3(TEXTC / 128, (BATCH * NTXT) / 64, 1), blk, 0, stream>>>(
        o1h, WoT, bo, oOut, TEXTC, TEXTC, TEXTC, TEXTC, 0, 0, 0, 0, 0, 0, 0);
}
